// AdditiveAttention_77395310674348
// MI455X (gfx1250) — matrix operand order repaired and checked by reading, not compiled, not run on hardware
//
#include <hip/hip_runtime.h>
#include <math.h>

#define DIM   1024
#define BATCH 32
#define SEQ   2048

typedef __bf16 bf16_t;
typedef __attribute__((ext_vector_type(16))) __bf16     v16bf;
typedef __attribute__((ext_vector_type(8)))  __bf16     v8bf;
typedef __attribute__((ext_vector_type(4)))  __bf16     bf16x4;
typedef __attribute__((ext_vector_type(8)))  float      v8f;
typedef __attribute__((ext_vector_type(4)))  float      f32x4;

union V16U { v16bf v; v8bf h[2]; };

// ---------------------------------------------------------------------------
// K1: q_proj[b,e] = dot(query[b,:], W1_w[e,:]) + W1_b[e]     (32x1024 outputs)
// ---------------------------------------------------------------------------
__global__ __launch_bounds__(256) void qproj_kernel(
    const float* __restrict__ query, const float* __restrict__ w1,
    const float* __restrict__ b1, float* __restrict__ out)
{
    int idx = blockIdx.x * 256 + threadIdx.x;      // 0 .. 32767
    int b = idx >> 10, e = idx & (DIM - 1);
    const f32x4* qr = (const f32x4*)(query + (size_t)b * DIM);
    const f32x4* wr = (const f32x4*)(w1 + (size_t)e * DIM);
    float a0 = 0.f, a1 = 0.f, a2 = 0.f, a3 = 0.f;
#pragma unroll 4
    for (int i = 0; i < DIM / 4; ++i) {
        f32x4 x = qr[i], w = wr[i];
        a0 += x.x * w.x; a1 += x.y * w.y; a2 += x.z * w.z; a3 += x.w * w.w;
    }
    out[idx] = (a0 + a1) + (a2 + a3) + b1[e];
}

// ---------------------------------------------------------------------------
// K2: convert W2_w (f32, row-major [e][d]) to bf16 (same layout).
// Column e of the GEMM B-matrix is then contiguous over d.
// ---------------------------------------------------------------------------
__global__ __launch_bounds__(256) void w2cvt_kernel(
    const float* __restrict__ w2, bf16_t* __restrict__ out)
{
    size_t idx = ((size_t)blockIdx.x * 256 + threadIdx.x) * 4;
    f32x4 v = *(const f32x4*)(w2 + idx);
    bf16x4 o;
    o.x = (bf16_t)v.x; o.y = (bf16_t)v.y; o.z = (bf16_t)v.z; o.w = (bf16_t)v.w;
    *(bf16x4*)(out + idx) = o;
}

// ---------------------------------------------------------------------------
// K3: fused  score[b,s] = V . tanh(key@W2^T + W2_b + q_proj[b]) + V_b
// One block = one 16-row M-tile (rows share the same batch b since S%16==0).
// 8 waves x 8 N-tiles each; K-loop with v_wmma_f32_16x16x32_bf16.
// ---------------------------------------------------------------------------
#define APAD 1032   // bf16 row stride: 2064B = 16B aligned, 4-dword bank step

__global__ __launch_bounds__(256) void score_kernel(
    const float*  __restrict__ key,
    const bf16_t* __restrict__ w2bf,
    const float*  __restrict__ qproj,
    const float*  __restrict__ w2b,
    const float*  __restrict__ vw,
    const float*  __restrict__ vb,
    float*        __restrict__ score_out)
{
    __shared__ bf16_t aTile[16][APAD];
    __shared__ float  sred[8][16];

    const int tid  = threadIdx.x;
    const int wave = tid >> 5;
    const int lane = tid & 31;
    const int half = lane >> 4;     // 0: lanes 0-15, 1: lanes 16-31
    const int lr   = lane & 15;

    const int    row0    = blockIdx.x * 16;   // global row = b*SEQ + s
    const int    b       = row0 >> 11;        // SEQ == 2048
    const size_t keyBase = (size_t)row0 * DIM;

    // ---- stage 16x1024 f32 key rows into LDS as bf16 ----
#pragma unroll
    for (int it = 0; it < 16; ++it) {
        int idx = (it * 256 + tid) * 4;       // 16384 floats total
        int r = idx >> 10, c = idx & (DIM - 1);
        f32x4 v = *(const f32x4*)(key + keyBase + (size_t)r * DIM + c);
        bf16x4 o;
        o.x = (bf16_t)v.x; o.y = (bf16_t)v.y; o.z = (bf16_t)v.z; o.w = (bf16_t)v.w;
        *(bf16x4*)&aTile[r][c] = o;
    }
    __syncthreads();

    // ---- WMMA main loop: this wave owns N-tiles nbBase .. nbBase+127 ----
    const int nbBase = wave * 128;
    v8f acc[8] = {};

    const bf16_t* arow = &aTile[lr][0];
    const int aOff = half * 8;                // A layout: halves split K by 8

    for (int kb = 0; kb < DIM; kb += 32) {
        V16U a;
        a.h[0] = *(const v8bf*)(arow + kb + aOff);        // K kb+0..7  / +8..15
        a.h[1] = *(const v8bf*)(arow + kb + aOff + 16);   // K kb+16..23/ +24..31
#pragma unroll
        for (int j = 0; j < 8; ++j) {
            // B layout: lane n holds column nb+n, K split by 8 per half (same as A)
            const bf16_t* bcol =
                w2bf + (size_t)(nbBase + j * 16 + lr) * DIM + kb + half * 8;
            V16U bf;
            bf.h[0] = *(const v8bf*)(bcol);
            bf.h[1] = *(const v8bf*)(bcol + 16);
            acc[j] = __builtin_amdgcn_wmma_f32_16x16x32_bf16(
                false, a.v, false, bf.v, (short)0, acc[j], false, false);
        }
    }

    // ---- epilogue: +bias +q_proj, tanh, dot with V, reduce over N ----
    float sacc[8] = {0.f, 0.f, 0.f, 0.f, 0.f, 0.f, 0.f, 0.f};
#pragma unroll
    for (int j = 0; j < 8; ++j) {
        const int col = nbBase + j * 16 + lr;   // N index held by this lane
        const float bq  = qproj[b * DIM + col] + w2b[col];
        const float vwv = vw[col];
#pragma unroll
        for (int v = 0; v < 8; ++v)             // C layout: VGPR v -> M = v+8*half
            sacc[v] += vwv * tanhf(acc[j][v] + bq);
    }

    // reduce across the 16 lanes of each half (xor masks stay within half)
#pragma unroll
    for (int v = 0; v < 8; ++v) {
        float s = sacc[v];
        s += __shfl_xor(s, 1, 32);
        s += __shfl_xor(s, 2, 32);
        s += __shfl_xor(s, 4, 32);
        s += __shfl_xor(s, 8, 32);
        if (lr == 0) sred[wave][half * 8 + v] = s;   // M = half*8 + v
    }
    __syncthreads();

    // sum the 8 waves' disjoint N-ranges, add V_b, emit raw scores
    if (tid < 16) {
        float t = vb[0];
#pragma unroll
        for (int w = 0; w < 8; ++w) t += sred[w][tid];
        score_out[row0 + tid] = t;
    }
}

// ---------------------------------------------------------------------------
// K4: softmax over S=2048, in place, one block per batch row
// ---------------------------------------------------------------------------
__global__ __launch_bounds__(256) void softmax_kernel(float* __restrict__ attn)
{
    __shared__ float red[256];
    const int t = threadIdx.x;
    float* row = attn + (size_t)blockIdx.x * SEQ;

    float vals[8];
    float m = -INFINITY;
#pragma unroll
    for (int i = 0; i < 8; ++i) { vals[i] = row[t + i * 256]; m = fmaxf(m, vals[i]); }
    red[t] = m; __syncthreads();
    for (int off = 128; off > 0; off >>= 1) {
        if (t < off) red[t] = fmaxf(red[t], red[t + off]);
        __syncthreads();
    }
    m = red[0]; __syncthreads();

    float s = 0.f;
#pragma unroll
    for (int i = 0; i < 8; ++i) { vals[i] = __expf(vals[i] - m); s += vals[i]; }
    red[t] = s; __syncthreads();
    for (int off = 128; off > 0; off >>= 1) {
        if (t < off) red[t] += red[t + off];
        __syncthreads();
    }
    const float inv = 1.0f / red[0];
#pragma unroll
    for (int i = 0; i < 8; ++i) row[t + i * 256] = vals[i] * inv;
}

// ---------------------------------------------------------------------------
// K5: context[b,d] = sum_s attn[b,s] * value[b,s,d]   (streams value once)
// ---------------------------------------------------------------------------
__global__ __launch_bounds__(256) void context_kernel(
    const float* __restrict__ value, const float* __restrict__ attn,
    float* __restrict__ ctx)
{
    __shared__ float aSh[SEQ];
    const int b = blockIdx.y;
    const int d = blockIdx.x * 256 + threadIdx.x;
    for (int i = threadIdx.x; i < SEQ; i += 256)
        aSh[i] = attn[(size_t)b * SEQ + i];
    __syncthreads();

    const float* vp = value + (size_t)b * SEQ * DIM + d;
    float a0 = 0.f, a1 = 0.f, a2 = 0.f, a3 = 0.f;
    for (int s = 0; s < SEQ; s += 4) {
        a0 += aSh[s + 0] * vp[(size_t)(s + 0) * DIM];
        a1 += aSh[s + 1] * vp[(size_t)(s + 1) * DIM];
        a2 += aSh[s + 2] * vp[(size_t)(s + 2) * DIM];
        a3 += aSh[s + 3] * vp[(size_t)(s + 3) * DIM];
    }
    ctx[(size_t)b * DIM + d] = (a0 + a1) + (a2 + a3);
}

// ---------------------------------------------------------------------------
extern "C" void kernel_launch(void* const* d_in, const int* in_sizes, int n_in,
                              void* d_out, int out_size, void* d_ws, size_t ws_size,
                              hipStream_t stream)
{
    const float* query = (const float*)d_in[0];
    const float* key   = (const float*)d_in[1];
    const float* value = (const float*)d_in[2];
    const float* W1w   = (const float*)d_in[3];
    const float* W1b   = (const float*)d_in[4];
    const float* W2w   = (const float*)d_in[5];
    const float* W2b   = (const float*)d_in[6];
    const float* Vw    = (const float*)d_in[7];
    const float* Vb    = (const float*)d_in[8];

    float* out  = (float*)d_out;
    float* ctx  = out;                     // [32,1024]
    float* attn = out + BATCH * DIM;       // [32,2048]

    float*  qproj = (float*)d_ws;                                   // 128 KB
    bf16_t* w2bf  = (bf16_t*)((char*)d_ws + (size_t)BATCH * DIM * sizeof(float)); // 2 MB

    qproj_kernel  <<<BATCH * DIM / 256, 256, 0, stream>>>(query, W1w, W1b, qproj);
    w2cvt_kernel  <<<DIM * DIM / 4 / 256, 256, 0, stream>>>(W2w, w2bf);
    score_kernel  <<<BATCH * SEQ / 16, 256, 0, stream>>>(key, w2bf, qproj,
                                                         W2b, Vw, Vb, attn);
    softmax_kernel<<<BATCH, 256, 0, stream>>>(attn);
    context_kernel<<<dim3(DIM / 256, BATCH), 256, 0, stream>>>(value, attn, ctx);
}
